// SymplecticGating_75746043232431
// MI455X (gfx1250) — compile-verified, matrix operand order checked by reading, not run on hardware
//
#include <hip/hip_runtime.h>
#include <hip/hip_bf16.h>

// ---------------- problem constants ----------------
#define BB 4
#define SS 4096
#define DD 1024
#define BS (BB * SS)      // 16384 tokens
#define CB 16             // codebook size
#define LDA 40            // LDS row stride (bf16 elems): 64B row + 16B TDM pad

typedef __bf16 bf16_t;
typedef __attribute__((ext_vector_type(8)))  bf16_t v8bf;
typedef __attribute__((ext_vector_type(16))) bf16_t v16bf;
typedef __attribute__((ext_vector_type(8)))  float  v8f;

#if defined(__has_builtin)
#if __has_builtin(__builtin_amdgcn_tensor_load_to_lds)
#define USE_TDM 1
#endif
#endif

#ifdef USE_TDM
typedef unsigned int u32x4 __attribute__((ext_vector_type(4)));
typedef int          i32x8 __attribute__((ext_vector_type(8)));
typedef int          i32x4 __attribute__((ext_vector_type(4)));

// generic pointer to LDS: low 32 bits are the LDS byte offset
__device__ __forceinline__ unsigned lds_off(const void* p) {
  return (unsigned)(unsigned long long)(uintptr_t)p;
}

// Issue a TDM 2D tile load: tile (tile0 x tile1) bf16 elements from a
// row-major tensor (row stride = stride0 elems) into LDS at lds_addr,
// padded +16B every 64B so LDS row stride = 80B (= LDA bf16 elems).
__device__ __forceinline__ void tdm_load_2d(unsigned lds_addr, const void* gaddr,
                                            unsigned tile0, unsigned tile1,
                                            unsigned dim0, unsigned dim1,
                                            unsigned stride0) {
  unsigned long long ga = (unsigned long long)(uintptr_t)gaddr;
  u32x4 g0;
  g0[0] = 1u;                                          // count=1, user mode
  g0[1] = lds_addr;                                    // D# lds_addr
  g0[2] = (unsigned)(ga & 0xffffffffu);                // global_addr[31:0]
  g0[3] = (unsigned)((ga >> 32) & 0x1ffffffu)          // global_addr[56:32]
        | (2u << 30);                                  // type=2 (image)
  i32x8 g1;
  g1[0] = (int)((1u << 16)                             // data_size = 2 bytes
              | (1u << 20)                             // pad_enable
              | (3u << 22)                             // pad_interval: 16 DWORDs (64B)
              | (3u << 25));                           // pad_amount: 4 DWORDs (16B)
  g1[1] = (int)((dim0 & 0xffffu) << 16);               // tensor_dim0[15:0]
  g1[2] = (int)((dim0 >> 16) | ((dim1 & 0xffffu) << 16));
  g1[3] = (int)((dim1 >> 16) | (tile0 << 16));         // tile_dim0
  g1[4] = (int)(tile1 & 0xffffu);                      // tile_dim1 (tile_dim2=0)
  g1[5] = (int)stride0;                                // tensor_dim0_stride[31:0]
  g1[6] = 0;
  g1[7] = 0;
  i32x4 z4 = {0, 0, 0, 0};
  i32x8 z8 = {0, 0, 0, 0, 0, 0, 0, 0};
  __builtin_amdgcn_tensor_load_to_lds(g0, g1, z4, z4, z8, 0);
}
#endif

// Build a 16x32 (bf16) WMMA fragment register block from LDS:
// low 8 elems = K block at p[0..7], high 8 = K block at p[16..23].
__device__ __forceinline__ v16bf frag16(const bf16_t* p) {
  v8bf lo = *(const v8bf*)p;
  v8bf hi = *(const v8bf*)(p + 16);
  return __builtin_shufflevector(lo, hi, 0, 1, 2, 3, 4, 5, 6, 7,
                                         8, 9, 10, 11, 12, 13, 14, 15);
}

// order-preserving float -> uint map (for radix-select of top-k thresholds)
__device__ __forceinline__ unsigned ford(float f) {
  unsigned u = __float_as_uint(f);
  return (u & 0x80000000u) ? ~u : (u | 0x80000000u);
}

// ---------------- K1: fp32 -> bf16 convert ----------------
__global__ void cvt_bf16(const float* __restrict__ src, bf16_t* __restrict__ dst, int n) {
  int i = blockIdx.x * blockDim.x + threadIdx.x;
  int stride = gridDim.x * blockDim.x;
  for (; i < n; i += stride) dst[i] = (bf16_t)src[i];
}

// ---------------- K2: fused 4-projection GEMM (bf16 WMMA, fp32 accum) ---------
// Y = X @ W[proj].T ; proj: 0=q, 1=k, 2=gate_pre(+bg), 3=mag(relu(.+bm))
// Tile feed: Tensor Data Mover (TDM) async loads into double-buffered LDS,
// synchronized with TENSORcnt + workgroup barriers.
__global__ __launch_bounds__(256)
void gemm_proj(const bf16_t* __restrict__ X, const bf16_t* __restrict__ W4,
               const float* __restrict__ bg, const float* __restrict__ bm,
               float* __restrict__ Qo, float* __restrict__ Ko,
               float* __restrict__ Go, float* __restrict__ Mo) {
  __shared__ __align__(16) bf16_t As[2][128 * LDA];
  __shared__ __align__(16) bf16_t Bs[2][128 * LDA];

  const int tid  = threadIdx.x;
  const int lane = tid & 31;
  const int wid  = tid >> 5;
  const int wm   = wid & 3;    // wave M slab (32 rows)
  const int wn   = wid >> 2;   // wave N slab (64 cols)
  const int lrow = lane & 15;
  const int half = lane >> 4;

  const int rowBase = blockIdx.x * 128;
  const int colBase = blockIdx.y * 128;
  const int proj    = blockIdx.z;
  const bf16_t* Wp  = W4 + (size_t)proj * DD * DD;

  const bf16_t* gA = X  + (size_t)rowBase * DD;   // A tile origin (k added per slab)
  const bf16_t* gB = Wp + (size_t)colBase * DD;   // B tile origin

  v8f acc[2][4];
#pragma unroll
  for (int mt = 0; mt < 2; ++mt)
#pragma unroll
    for (int nt = 0; nt < 4; ++nt)
#pragma unroll
      for (int j = 0; j < 8; ++j) acc[mt][nt][j] = 0.0f;

#ifdef USE_TDM
  const unsigned ldsA[2] = { lds_off(&As[0][0]), lds_off(&As[1][0]) };
  const unsigned ldsB[2] = { lds_off(&Bs[0][0]), lds_off(&Bs[1][0]) };

  if (wid == 0) {   // one wave drives the DMA engine (EXEC ignored by TDM)
    tdm_load_2d(ldsA[0], gA + 0,  32, 128, DD, BS, DD);
    tdm_load_2d(ldsB[0], gB + 0,  32, 128, DD, DD, DD);
    tdm_load_2d(ldsA[1], gA + 32, 32, 128, DD, BS, DD);
    tdm_load_2d(ldsB[1], gB + 32, 32, 128, DD, DD, DD);
  }

  for (int ks = 0; ks < DD / 32; ++ks) {
    const int cur = ks & 1;
    if (wid == 0) {
      if (ks < DD / 32 - 1) __builtin_amdgcn_s_wait_tensorcnt(2);  // pair ks done
      else                  __builtin_amdgcn_s_wait_tensorcnt(0);  // last pair done
    }
    __syncthreads();   // publish TDM-written LDS to all waves

    v16bf af[2], bf[4];
#pragma unroll
    for (int mt = 0; mt < 2; ++mt)
      af[mt] = frag16(&As[cur][(wm * 32 + mt * 16 + lrow) * LDA + half * 8]);
#pragma unroll
    for (int nt = 0; nt < 4; ++nt)
      bf[nt] = frag16(&Bs[cur][(wn * 64 + nt * 16 + lrow) * LDA + half * 8]);

#pragma unroll
    for (int mt = 0; mt < 2; ++mt)
#pragma unroll
      for (int nt = 0; nt < 4; ++nt)
        acc[mt][nt] = __builtin_amdgcn_wmma_f32_16x16x32_bf16(
            false, af[mt], false, bf[nt], (short)0, acc[mt][nt], false, false);

    __syncthreads();   // all waves done reading buf[cur]; safe to refill it
    if (ks + 2 < DD / 32 && wid == 0) {
      const int k0 = (ks + 2) * 32;
      tdm_load_2d(ldsA[cur], gA + k0, 32, 128, DD, BS, DD);
      tdm_load_2d(ldsB[cur], gB + k0, 32, 128, DD, DD, DD);
    }
  }
#else
  // Fallback: global->reg->LDS staging
  const int srow = tid >> 1;
  const int scol = (tid & 1) * 16;
  const bf16_t* ga = X  + (size_t)(rowBase + srow) * DD + scol;
  const bf16_t* gb = Wp + (size_t)(colBase + srow) * DD + scol;

  uint4 ra0, ra1, rb0, rb1;
  auto loadRegs = [&](int k0) {
    const uint4* pa = (const uint4*)(ga + k0);
    const uint4* pb = (const uint4*)(gb + k0);
    ra0 = pa[0]; ra1 = pa[1];
    rb0 = pb[0]; rb1 = pb[1];
    if (k0 + 64 < DD) {
      __builtin_prefetch(ga + k0 + 64, 0, 1);
      __builtin_prefetch(gb + k0 + 64, 0, 1);
    }
  };
  auto storeLds = [&](int buf) {
    uint4* da = (uint4*)&As[buf][srow * LDA + scol];
    uint4* db = (uint4*)&Bs[buf][srow * LDA + scol];
    da[0] = ra0; da[1] = ra1;
    db[0] = rb0; db[1] = rb1;
  };

  loadRegs(0);
  storeLds(0);
  __syncthreads();
  for (int ks = 0; ks < DD / 32; ++ks) {
    const int cur = ks & 1;
    if (ks + 1 < DD / 32) loadRegs((ks + 1) * 32);

    v16bf af[2], bf[4];
#pragma unroll
    for (int mt = 0; mt < 2; ++mt)
      af[mt] = frag16(&As[cur][(wm * 32 + mt * 16 + lrow) * LDA + half * 8]);
#pragma unroll
    for (int nt = 0; nt < 4; ++nt)
      bf[nt] = frag16(&Bs[cur][(wn * 64 + nt * 16 + lrow) * LDA + half * 8]);
#pragma unroll
    for (int mt = 0; mt < 2; ++mt)
#pragma unroll
      for (int nt = 0; nt < 4; ++nt)
        acc[mt][nt] = __builtin_amdgcn_wmma_f32_16x16x32_bf16(
            false, af[mt], false, bf[nt], (short)0, acc[mt][nt], false, false);

    if (ks + 1 < DD / 32) {
      __syncthreads();
      storeLds(cur ^ 1);
      __syncthreads();
    }
  }
#endif

  float* outp = (proj == 0) ? Qo : (proj == 1) ? Ko : (proj == 2) ? Go : Mo;
#pragma unroll
  for (int mt = 0; mt < 2; ++mt)
#pragma unroll
    for (int nt = 0; nt < 4; ++nt) {
      int col = colBase + wn * 64 + nt * 16 + lrow;
      float bias = (proj == 2) ? bg[col] : (proj == 3) ? bm[col] : 0.0f;
#pragma unroll
      for (int j = 0; j < 8; ++j) {
        int row = rowBase + wm * 32 + mt * 16 + half * 8 + j;
        float v = acc[mt][nt][j] + bias;
        if (proj == 3) v = fmaxf(v, 0.0f);
        outp[(size_t)row * DD + col] = v;
      }
    }
}

// ---------------- K3: codebook logits + softmax + top-4 renorm --------------
__global__ __launch_bounds__(256)
void codebook_probs(const bf16_t* __restrict__ X, const bf16_t* __restrict__ Wc,
                    float* __restrict__ probs) {
  __shared__ bf16_t wcl[CB * 1032];   // padded rows: bank-conflict-free
  __shared__ bf16_t xl[8 * DD];
  const int tid = threadIdx.x, lane = tid & 31, w = tid >> 5;

  for (int i = tid; i < CB * DD; i += 256) {
    int c = i >> 10, d = i & 1023;
    wcl[c * 1032 + d] = Wc[i];
  }
  const int tok = blockIdx.x * 8 + w;
  for (int d = lane; d < DD; d += 32) xl[w * DD + d] = X[(size_t)tok * DD + d];
  __syncthreads();

  const int c = lane & 15, h = lane >> 4;   // lanes c / c+16 split halves of D
  float acc = 0.0f;
  for (int d = h * 512; d < h * 512 + 512; ++d)
    acc += (float)xl[w * DD + d] * (float)wcl[c * 1032 + d];
  acc += __shfl_xor(acc, 16, 32);

  float logit = acc * 10.0f;                // / CB_TEMP (0.1)
  float mx = logit;
  for (int o = 8; o > 0; o >>= 1) mx = fmaxf(mx, __shfl_xor(mx, o, 16));
  float e = __expf(logit - mx);
  float s = e;
  for (int o = 8; o > 0; o >>= 1) s += __shfl_xor(s, o, 16);
  float p = e / s;

  int r = 0;
  for (int j = 0; j < 16; ++j) {
    float pj = __shfl(p, j, 16);
    r += (pj > p) || (pj == p && j < c);
  }
  float ps = (r < 4) ? p : 0.0f;
  float s2 = ps;
  for (int o = 8; o > 0; o >>= 1) s2 += __shfl_xor(s2, o, 16);
  float pr = ps / fmaxf(s2, 1e-8f);
  if (lane < 16) probs[(size_t)tok * CB + c] = pr;
}

// ---------------- K4: adaptive top-k gate + twist + complexity --------------
__global__ __launch_bounds__(256)
void gate_twist(const float* __restrict__ Q, const float* __restrict__ K,
                const float* __restrict__ G, const float* __restrict__ M,
                float* __restrict__ comp) {
  __shared__ float smf[8];
  __shared__ int   smi[8];
  __shared__ int   bkk;
  __shared__ int   bcnt;
  const int tid = threadIdx.x, lane = tid & 31, w = tid >> 5;
  const int t = blockIdx.x;
  const int s = t & (SS - 1);

  float g[4]; unsigned mv[4];
  float sabs = 0.0f;
#pragma unroll
  for (int e = 0; e < 4; ++e) {
    int i = tid + e * 256;
    g[e] = G[(size_t)t * DD + i];
    mv[e] = ford(g[e]);
    sabs += fabsf(g[e]);
  }
  for (int o = 16; o > 0; o >>= 1) sabs += __shfl_down(sabs, o, 32);
  if (lane == 0) smf[w] = sabs;
  __syncthreads();
  if (tid == 0) {
    float v = 0.0f;
    for (int i = 0; i < 8; ++i) v += smf[i];
    float strength = tanhf(v * (1.0f / (float)DD));
    int kk = (int)ceilf(strength * 256.0f);   // ADAPT_RATIO * D
    bkk = min(max(kk, 1), DD);
  }
  __syncthreads();
  const int kk = bkk;

  // radix-select the kk-th largest gate_pre value (uint-ordered)
  unsigned thr = 0;
  for (int bit = 31; bit >= 0; --bit) {
    unsigned cand = thr | (1u << bit);
    int cnt = 0;
#pragma unroll
    for (int e = 0; e < 4; ++e) cnt += (mv[e] >= cand) ? 1 : 0;
    for (int o = 16; o > 0; o >>= 1) cnt += __shfl_down(cnt, o, 32);
    if (lane == 0) smi[w] = cnt;
    __syncthreads();
    if (tid == 0) {
      int c2 = 0;
      for (int i = 0; i < 8; ++i) c2 += smi[i];
      bcnt = c2;
    }
    __syncthreads();
    if (bcnt >= kk) thr = cand;
  }

  float sum = 0.0f;
#pragma unroll
  for (int e = 0; e < 4; ++e) {
    int i = tid + e * 256;
    float gate = (g[e] > 0.0f && mv[e] >= thr) ? 1.0f : 0.0f;
    float tw = 0.0f;
    if (s > 0) {
      float qp = Q[(size_t)(t - 1) * DD + i], kp = K[(size_t)(t - 1) * DD + i];
      float qc = Q[(size_t)t * DD + i],       kc = K[(size_t)t * DD + i];
      tw = qp * kc - qc * kp;
    }
    float mu = gate * M[(size_t)t * DD + i] * tw;
    sum += fabsf(mu);
  }
  for (int o = 16; o > 0; o >>= 1) sum += __shfl_down(sum, o, 32);
  if (lane == 0) smf[w] = sum;
  __syncthreads();
  if (tid == 0) {
    float v = 0.0f;
    for (int i = 0; i < 8; ++i) v += smf[i];
    comp[t] = v * (1.0f / (float)DD);
  }
}

// ---------------- K5: quorum smoothing + budgeted (top-2048) mask -----------
__global__ __launch_bounds__(256)
void quorum_budget(const float* __restrict__ comp, float* __restrict__ comp2) {
  __shared__ int smi[8];
  __shared__ int bcnt;
  const int tid = threadIdx.x, lane = tid & 31, w = tid >> 5;
  const float* c = comp + (size_t)blockIdx.x * SS;

  float cv[16], qv[16]; unsigned mv[16];
#pragma unroll
  for (int e = 0; e < 16; ++e) {
    int s = tid + e * 256;
    float c0 = (s > 0) ? c[s - 1] : 0.0f;
    float c1 = c[s];
    float c2 = (s < SS - 1) ? c[s + 1] : 0.0f;
    float local = (c0 + c1 + c2) * (1.0f / 3.0f);                 // Q_WIN=3
    float q = 1.0f / (1.0f + __expf(-(local - 0.5f) * 10.0f));    // Q_THR/Q_TEMP
    cv[e] = c1; qv[e] = q; mv[e] = ford(q);
  }
  unsigned thr = 0;
  for (int bit = 31; bit >= 0; --bit) {
    unsigned cand = thr | (1u << bit);
    int cnt = 0;
#pragma unroll
    for (int e = 0; e < 16; ++e) cnt += (mv[e] >= cand) ? 1 : 0;
    for (int o = 16; o > 0; o >>= 1) cnt += __shfl_down(cnt, o, 32);
    if (lane == 0) smi[w] = cnt;
    __syncthreads();
    if (tid == 0) {
      int c2 = 0;
      for (int i = 0; i < 8; ++i) c2 += smi[i];
      bcnt = c2;
    }
    __syncthreads();
    if (bcnt >= 2048) thr = cand;   // BUDGET_K
  }
#pragma unroll
  for (int e = 0; e < 16; ++e) {
    int s = tid + e * 256;
    float qm = (mv[e] >= thr) ? qv[e] : 0.0f;
    comp2[(size_t)blockIdx.x * SS + s] = cv[e] * (0.5f + 0.5f * qm);  // Q_MIX
  }
}

// ---------------- K6: codebook temporal shift + final mix -------------------
__global__ void final_mix(const float* __restrict__ comp2,
                          const float* __restrict__ probs,
                          float* __restrict__ out) {
  int t = blockIdx.x * blockDim.x + threadIdx.x;
  if (t >= BS) return;
  int s = t & (SS - 1);
  float cb = 0.0f;
  if (s > 0) {
    float acc = 0.0f;
#pragma unroll
    for (int ci = 0; ci < CB; ++ci)
      acc += fabsf(probs[(size_t)t * CB + ci] - probs[(size_t)(t - 1) * CB + ci]);
    cb = 0.5f * acc;
  }
  out[t] = 0.7f * comp2[t] + 0.3f * cb;   // (1-CB_MIX), CB_MIX
}

// ---------------- launch ----------------
extern "C" void kernel_launch(void* const* d_in, const int* in_sizes, int n_in,
                              void* d_out, int out_size, void* d_ws, size_t ws_size,
                              hipStream_t stream) {
  const float* x  = (const float*)d_in[0];
  const float* Wq = (const float*)d_in[1];
  const float* Wk = (const float*)d_in[2];
  const float* Wg = (const float*)d_in[3];
  const float* bg = (const float*)d_in[4];
  const float* Wm = (const float*)d_in[5];
  const float* bm = (const float*)d_in[6];
  const float* Wc = (const float*)d_in[7];
  float* out = (float*)d_out;

  // workspace layout: xb 32MB | wb 8MB | wcb 32KB | q/k/g/m 4x64MB |
  // probs 1MB | comp 64KB | comp2 64KB   (~300MB total)
  bf16_t* xb  = (bf16_t*)d_ws;
  bf16_t* wb  = xb + (size_t)BS * DD;
  bf16_t* wcb = wb + 4ull * DD * DD;
  float* qb = (float*)(wcb + (size_t)CB * DD);
  float* kb = qb + (size_t)BS * DD;
  float* gb = kb + (size_t)BS * DD;
  float* mb = gb + (size_t)BS * DD;
  float* pb = mb + (size_t)BS * DD;
  float* c1 = pb + (size_t)BS * CB;
  float* c2 = c1 + BS;

  cvt_bf16<<<1024, 256, 0, stream>>>(x, xb, BS * DD);
  cvt_bf16<<<512, 256, 0, stream>>>(Wq, wb + 0ull * DD * DD, DD * DD);
  cvt_bf16<<<512, 256, 0, stream>>>(Wk, wb + 1ull * DD * DD, DD * DD);
  cvt_bf16<<<512, 256, 0, stream>>>(Wg, wb + 2ull * DD * DD, DD * DD);
  cvt_bf16<<<512, 256, 0, stream>>>(Wm, wb + 3ull * DD * DD, DD * DD);
  cvt_bf16<<<64, 256, 0, stream>>>(Wc, wcb, CB * DD);

  dim3 gg(BS / 128, DD / 128, 4);
  gemm_proj<<<gg, 256, 0, stream>>>(xb, wb, bg, bm, qb, kb, gb, mb);
  codebook_probs<<<BS / 8, 256, 0, stream>>>(xb, wcb, pb);
  gate_twist<<<BS, 256, 0, stream>>>(qb, kb, gb, mb, c1);
  quorum_budget<<<BB, 256, 0, stream>>>(c1, c2);
  final_mix<<<(BS + 255) / 256, 256, 0, stream>>>(c2, pb, out);
}
